// VQVAE_9517647528584
// MI455X (gfx1250) — matrix operand order repaired and checked by reading, not compiled, not run on hardware
//
#include <hip/hip_runtime.h>
#include <hip/hip_fp16.h>

typedef __attribute__((ext_vector_type(16))) _Float16 v16h;
typedef __attribute__((ext_vector_type(8)))  float    v8f;

#define NROWS   131072
#define INDIM   64
#define EMBD    64
#define KCODES  1024
#define DENSE   256
#define ASTRIDE 264   // 256 + 8 halfs pad: rows land on different LDS banks
#define MROWS   32    // rows per wave: each B fragment feeds 2 WMMAs

// ---- packed-weight workspace layout (offsets in halfs) ----
// B-fragment order: frag (nt,kt) -> 32 lanes * 16 halfs contiguous.
// lane<16 : n = nt*16+lane,    halves j=0..15 hold k = kt*32 + j
// lane>=16: n = nt*16+lane-16, halves j=0..15 hold k = kt*32 + 16 + j
static constexpr int OFF_WE1    = 0;                       // 256x64
static constexpr int OFF_WE2    = OFF_WE1 + 256 * 64;      // 256x256
static constexpr int OFF_WE3    = OFF_WE2 + 256 * 256;     // 64x256
static constexpr int OFF_EMBB   = OFF_WE3 + 64 * 256;      // 1024x64 (B for z@emb^T)
static constexpr int OFF_WD1    = OFF_EMBB + 1024 * 64;    // 256x64
static constexpr int OFF_WD2    = OFF_WD1 + 256 * 64;      // 256x256
static constexpr int OFF_WD3    = OFF_WD2 + 256 * 256;     // 64x256
static constexpr int OFF_EMBROW = OFF_WD3 + 64 * 256;      // 1024x64 row-major f16
static constexpr int HALF_TOTAL = OFF_EMBROW + 1024 * 64;
static constexpr int ENORM_BYTE = HALF_TOTAL * 2;          // 1024 f32 after the halfs

// ---------------- prep kernels ----------------
__global__ void pack_weight(const float* __restrict__ W, _Float16* __restrict__ out,
                            int Nout, int Kin) {
  int t = blockIdx.x * 256 + threadIdx.x;
  if (t >= Nout * Kin) return;
  int KT   = Kin >> 5;
  int frag = t >> 9;              // 512 halfs per fragment
  int r    = t & 511;
  int lane = r >> 4;
  int j    = r & 15;
  int nt   = frag / KT;
  int kt   = frag - nt * KT;
  int n    = nt * 16 + (lane & 15);
  int k    = kt * 32 + ((lane >> 4) << 4) + j;
  out[t] = (_Float16)W[n * Kin + k];
}

__global__ void pack_row_norm(const float* __restrict__ emb,
                              _Float16* __restrict__ embRow,
                              float* __restrict__ enorm) {
  int n = blockIdx.x;        // 1024
  int t = threadIdx.x;       // 64
  float v = emb[n * 64 + t];
  embRow[n * 64 + t] = (_Float16)v;
  __shared__ float s[64];
  s[t] = v * v;
  __syncthreads();
  if (t < 32) s[t] += s[t + 32];
  __syncthreads();
  if (t == 0) {
    float acc = 0.f;
    #pragma unroll
    for (int i = 0; i < 32; ++i) acc += s[i];
    enorm[n] = acc;
  }
}

// ---------------- device helpers ----------------
// Branchless SELU: for x>0 the exp-term is exp(0)-1 == 0; for x<=0 the max-term is 0.
// 6 VALU ops, no EXEC divergence; v_exp_f32 is TRANS and co-executes with WMMA.
__device__ __forceinline__ float selu_f(float x) {
  const float s  = 1.0507009873554805f;
  const float sa = 1.7580993408473766f;       // scale * alpha
  return fmaf(s, fmaxf(x, 0.f), sa * (__expf(fminf(x, 0.f)) - 1.f));
}

// A fragment 16x32, contiguous k order used here (same as packed B): M = lane%16; lane<16: K = {0..15}; lane>=16: K = {16..31}
__device__ __forceinline__ v16h load_afrag(const _Float16* actp, int lane, int kt) {
  int m  = lane & 15;
  int lo = (lane >> 4) << 4;  // 0 or 16
  const _Float16* base = actp + m * ASTRIDE + kt * 32 + lo;
  union { v16h h; uint4 u[2]; } f;
  f.u[0] = *(const uint4*)(base);        // K = lo..lo+7
  f.u[1] = *(const uint4*)(base + 8);    // K = lo+8..lo+15
  return f.h;
}

__device__ __forceinline__ v16h load_bfrag(const _Float16* w, int lane, int nt, int kt, int KT) {
  const _Float16* base = w + (((size_t)(nt * KT + kt) * 32 + lane) << 4);
  union { v16h h; uint4 u[2]; } f;
  f.u[0] = *(const uint4*)(base);
  f.u[1] = *(const uint4*)(base + 8);
  return f.h;
}

#define WMMA_F16(A, B, C) \
  __builtin_amdgcn_wmma_f32_16x16x32_f16(false, (A), false, (B), (short)0, (C), false, false)

template <int KT, int NT, bool DO_SELU>
__device__ __forceinline__ void layer(const _Float16* __restrict__ wbase,
                                      const float* __restrict__ bias,
                                      _Float16* actlds, int lane) {
  v16h a[2][KT];
  #pragma unroll
  for (int t = 0; t < 2; ++t)
    #pragma unroll
    for (int kt = 0; kt < KT; ++kt)
      a[t][kt] = load_afrag(actlds + t * 16 * ASTRIDE, lane, kt);
  __syncthreads();                         // all lanes have A before overwriting act
  const int n0 = lane & 15;
  const int mb = (lane >> 4) << 3;
  for (int nt = 0; nt < NT; ++nt) {
    v8f c0 = {}, c1 = {};
    #pragma unroll
    for (int kt = 0; kt < KT; ++kt) {
      v16h b = load_bfrag(wbase, lane, nt, kt, KT);   // fetched once, used twice
      c0 = WMMA_F16(a[0][kt], b, c0);
      c1 = WMMA_F16(a[1][kt], b, c1);
    }
    const int n  = nt * 16 + n0;
    const float bv = bias[n];
    #pragma unroll
    for (int v = 0; v < 8; ++v) {
      float y0 = c0[v] + bv;
      float y1 = c1[v] + bv;
      if (DO_SELU) { y0 = selu_f(y0); y1 = selu_f(y1); }
      actlds[(mb + v) * ASTRIDE + n]        = (_Float16)y0;
      actlds[(16 + mb + v) * ASTRIDE + n]   = (_Float16)y1;
    }
  }
  __syncthreads();
}

template <int KT>
__device__ __forceinline__ void layer_out(const _Float16* __restrict__ wbase,
                                          const float* __restrict__ bias,
                                          const _Float16* actlds,
                                          float* __restrict__ out, int row0, int lane) {
  v16h a[2][KT];
  #pragma unroll
  for (int t = 0; t < 2; ++t)
    #pragma unroll
    for (int kt = 0; kt < KT; ++kt)
      a[t][kt] = load_afrag(actlds + t * 16 * ASTRIDE, lane, kt);
  const int n0 = lane & 15;
  const int mb = (lane >> 4) << 3;
  for (int nt = 0; nt < 4; ++nt) {         // Nout = 64
    v8f c0 = {}, c1 = {};
    #pragma unroll
    for (int kt = 0; kt < KT; ++kt) {
      v16h b = load_bfrag(wbase, lane, nt, kt, KT);
      c0 = WMMA_F16(a[0][kt], b, c0);
      c1 = WMMA_F16(a[1][kt], b, c1);
    }
    const int n  = nt * 16 + n0;
    const float bv = bias[n];
    #pragma unroll
    for (int v = 0; v < 8; ++v) {
      out[(size_t)(row0 + mb + v) * 64 + n]      = c0[v] + bv;
      out[(size_t)(row0 + 16 + mb + v) * 64 + n] = c1[v] + bv;
    }
  }
}

// ---------------- fused VQ-VAE kernel: 1 wave, 32 rows per block ----------------
__launch_bounds__(32)
__global__ void vqvae_fused(const float* __restrict__ x,
                            const _Float16* __restrict__ wpack,
                            const float* __restrict__ enorm,
                            const float* __restrict__ be1, const float* __restrict__ be2,
                            const float* __restrict__ be3, const float* __restrict__ bd1,
                            const float* __restrict__ bd2, const float* __restrict__ bd3,
                            float* __restrict__ recon, float* __restrict__ out_idx) {
  __shared__ _Float16 act[MROWS * ASTRIDE];
  __shared__ int sidx[MROWS];
  const int lane = threadIdx.x;
  const int row0 = blockIdx.x * MROWS;

  // stage x tile -> LDS as f16 (fully coalesced: 2048 contiguous f32)
  #pragma unroll
  for (int i = 0; i < 64; ++i) {
    int e = i * 32 + lane;
    act[(e >> 6) * ASTRIDE + (e & 63)] = (_Float16)x[(size_t)row0 * 64 + e];
  }
  __syncthreads();

  // encoder
  layer<2, 16, true >(wpack + OFF_WE1, be1, act, lane);   // 64  -> 256, selu
  layer<8, 16, true >(wpack + OFF_WE2, be2, act, lane);   // 256 -> 256, selu
  layer<8, 4,  false>(wpack + OFF_WE3, be3, act, lane);   // 256 -> 64 (z)

  // ---- quantize: argmin_n ( ||e_n||^2 - 2 z.e_n ) ----
  {
    v16h a[2][2];
    #pragma unroll
    for (int t = 0; t < 2; ++t) {
      a[t][0] = load_afrag(act + t * 16 * ASTRIDE, lane, 0);
      a[t][1] = load_afrag(act + t * 16 * ASTRIDE, lane, 1);
    }
    __syncthreads();
    const int n0 = lane & 15;
    const int mb = (lane >> 4) << 3;
    float bm[2][8];
    int   bi[2][8];
    #pragma unroll
    for (int t = 0; t < 2; ++t)
      #pragma unroll
      for (int v = 0; v < 8; ++v) { bm[t][v] = 3.0e38f; bi[t][v] = 0; }
    const _Float16* embB = wpack + OFF_EMBB;
    for (int nt = 0; nt < KCODES / 16; ++nt) {
      v16h b0 = load_bfrag(embB, lane, nt, 0, 2);
      v16h b1 = load_bfrag(embB, lane, nt, 1, 2);
      v8f c0 = {}, c1 = {};
      c0 = WMMA_F16(a[0][0], b0, c0);
      c0 = WMMA_F16(a[0][1], b1, c0);
      c1 = WMMA_F16(a[1][0], b0, c1);
      c1 = WMMA_F16(a[1][1], b1, c1);
      const int n  = nt * 16 + n0;
      const float en = enorm[n];
      #pragma unroll
      for (int v = 0; v < 8; ++v) {
        float d0 = fmaf(-2.f, c0[v], en);
        float d1 = fmaf(-2.f, c1[v], en);
        if (d0 < bm[0][v]) { bm[0][v] = d0; bi[0][v] = n; }
        if (d1 < bm[1][v]) { bm[1][v] = d1; bi[1][v] = n; }
      }
    }
    // reduce over the 16 lanes of each half-wave (ds_permute xor tree)
    #pragma unroll
    for (int msk = 8; msk >= 1; msk >>= 1) {
      #pragma unroll
      for (int t = 0; t < 2; ++t)
        #pragma unroll
        for (int v = 0; v < 8; ++v) {
          float ov = __shfl_xor(bm[t][v], msk, 32);
          int   oi = __shfl_xor(bi[t][v], msk, 32);
          if (ov < bm[t][v] || (ov == bm[t][v] && oi < bi[t][v])) {
            bm[t][v] = ov; bi[t][v] = oi;
          }
        }
    }
    if ((lane & 15) == 0) {
      #pragma unroll
      for (int t = 0; t < 2; ++t)
        #pragma unroll
        for (int v = 0; v < 8; ++v) sidx[t * 16 + mb + v] = bi[t][v];
    }
    __syncthreads();
    // straight-through: zq = emb[idx]; also emit idx (as float, after recon)
    {
      int id = sidx[lane];                 // one row per lane, all 32 lanes busy
      out_idx[row0 + lane] = (float)id;
      const uint4* src = (const uint4*)(wpack + OFF_EMBROW + (size_t)id * 64);
      uint4* dst = (uint4*)(&act[lane * ASTRIDE]);
      #pragma unroll
      for (int i = 0; i < 8; ++i) dst[i] = src[i];   // 64 halfs = 128 B
    }
    __syncthreads();
  }

  // decoder
  layer<2, 16, true>(wpack + OFF_WD1, bd1, act, lane);    // 64  -> 256, selu
  layer<8, 16, true>(wpack + OFF_WD2, bd2, act, lane);    // 256 -> 256, selu
  layer_out<8>(wpack + OFF_WD3, bd3, act, recon, row0, lane);  // 256 -> 64
}

// ---------------- launch ----------------
extern "C" void kernel_launch(void* const* d_in, const int* in_sizes, int n_in,
                              void* d_out, int out_size, void* d_ws, size_t ws_size,
                              hipStream_t stream) {
  const float* x   = (const float*)d_in[0];
  const float* We1 = (const float*)d_in[1];
  const float* be1 = (const float*)d_in[2];
  const float* We2 = (const float*)d_in[3];
  const float* be2 = (const float*)d_in[4];
  const float* We3 = (const float*)d_in[5];
  const float* be3 = (const float*)d_in[6];
  const float* emb = (const float*)d_in[7];
  const float* Wd1 = (const float*)d_in[8];
  const float* bd1 = (const float*)d_in[9];
  const float* Wd2 = (const float*)d_in[10];
  const float* bd2 = (const float*)d_in[11];
  const float* Wd3 = (const float*)d_in[12];
  const float* bd3 = (const float*)d_in[13];

  _Float16* wpack = (_Float16*)d_ws;
  float*    enorm = (float*)((char*)d_ws + ENORM_BYTE);

  // repack weights to f16 B-fragment order (cheap: <1 MB total, stays L2-resident)
  pack_weight<<<(256 * 64   + 255) / 256, 256, 0, stream>>>(We1, wpack + OFF_WE1, 256, 64);
  pack_weight<<<(256 * 256  + 255) / 256, 256, 0, stream>>>(We2, wpack + OFF_WE2, 256, 256);
  pack_weight<<<(64  * 256  + 255) / 256, 256, 0, stream>>>(We3, wpack + OFF_WE3, 64, 256);
  pack_weight<<<(1024 * 64  + 255) / 256, 256, 0, stream>>>(emb, wpack + OFF_EMBB, 1024, 64);
  pack_weight<<<(256 * 64   + 255) / 256, 256, 0, stream>>>(Wd1, wpack + OFF_WD1, 256, 64);
  pack_weight<<<(256 * 256  + 255) / 256, 256, 0, stream>>>(Wd2, wpack + OFF_WD2, 256, 256);
  pack_weight<<<(64  * 256  + 255) / 256, 256, 0, stream>>>(Wd3, wpack + OFF_WD3, 64, 256);
  pack_row_norm<<<1024, 64, 0, stream>>>(emb, wpack + OFF_EMBROW, enorm);

  float* recon = (float*)d_out;
  float* oidx  = (float*)d_out + (size_t)NROWS * INDIM;
  vqvae_fused<<<NROWS / MROWS, 32, 0, stream>>>(x, wpack, enorm,
                                                be1, be2, be3, bd1, bd2, bd3,
                                                recon, oidx);
}
